// MiniS4D_14422500179972
// MI455X (gfx1250) — compile-verified, matrix operand order checked by reading, not run on hardware
//
#include <hip/hip_runtime.h>
#include <hip/hip_bf16.h>

// ---------------------------------------------------------------------------
// MiniS4D on MI455X (gfx1250): wave32 + v_wmma_f32_16x16x32_f16 pipeline.
// ---------------------------------------------------------------------------

#define H_MODEL 512
#define N_STATE 64
#define SEQ_L   1024
#define NBATCH  8
#define O_MODEL 1024            // 2 * H_MODEL
#define KPAD    32              // leading zero taps for the Toeplitz band
#define KROW    (SEQ_L + KPAD)  // 1056 halves per channel row

typedef __attribute__((ext_vector_type(16))) _Float16 v16h;
typedef __attribute__((ext_vector_type(8)))  _Float16 v8h;
typedef __attribute__((ext_vector_type(8)))  float    v8f;

// ---------------------------------------------------------------------------
// Kernel 1: S4D kernel synthesis.  K[h,l] = Re( sum_n w[h,n] * exp(dtA*l) ).
// block = one channel h (64 threads). Threads first cooperatively compute the
// per-state constants into LDS, then each thread owns 16 output positions and
// runs the 64-state geometric recurrence privately (no reductions needed).
// Output: f16, padded with 32 leading zeros per row.
// ---------------------------------------------------------------------------
__global__ __launch_bounds__(64) void s4d_make_kernel(
    const float* __restrict__ log_dt, const float* __restrict__ log_A_real,
    const float* __restrict__ A_imag, const float* __restrict__ B_real,
    const float* __restrict__ B_imag, const float* __restrict__ C_real,
    const float* __restrict__ C_imag, _Float16* __restrict__ Kpad) {
  const int h = blockIdx.x;
  const int t = threadIdx.x;  // 0..63
  __shared__ float s_dAr[N_STATE], s_dAi[N_STATE], s_wr[N_STATE], s_wi[N_STATE];

  const float dt = expf(log_dt[h]);
  {
    const int n = t;
    const float Ar = -expf(log_A_real[h * N_STATE + n]);
    const float Ai = A_imag[h * N_STATE + n];
    const float dAr = Ar * dt, dAi = Ai * dt;
    // A_bar = exp(dtA)
    float sb, cb;
    sincosf(dAi, &sb, &cb);
    const float e = expf(dAr);
    const float abr = e * cb, abi = e * sb;
    // B_bar = (A_bar - 1) / A * B   (complex divide via conj(A)/|A|^2)
    const float nr = abr - 1.0f, ni = abi;
    const float den = Ar * Ar + Ai * Ai;
    const float ibr = (nr * Ar + ni * Ai) / den;
    const float ibi = (ni * Ar - nr * Ai) / den;
    const float Br = B_real[h * N_STATE + n], Bi = B_imag[h * N_STATE + n];
    const float bbr = ibr * Br - ibi * Bi;
    const float bbi = ibr * Bi + ibi * Br;
    const float Cr = C_real[h * N_STATE + n], Ci = C_imag[h * N_STATE + n];
    s_wr[n] = Cr * bbr - Ci * bbi;
    s_wi[n] = Cr * bbi + Ci * bbr;
    s_dAr[n] = dAr;
    s_dAi[n] = dAi;
  }
  __syncthreads();

  float acc[16];
#pragma unroll
  for (int j = 0; j < 16; ++j) acc[j] = 0.0f;
  const int l0 = t * 16;

  for (int n = 0; n < N_STATE; ++n) {
    const float dAr = s_dAr[n], dAi = s_dAi[n];
    const float wr = s_wr[n], wi = s_wi[n];
    // p = exp(dtA * l0)
    float sp, cp;
    sincosf(dAi * (float)l0, &sp, &cp);
    const float e0 = expf(dAr * (float)l0);
    float pr = e0 * cp, pi = e0 * sp;
    // step = exp(dtA)
    float s1, c1;
    sincosf(dAi, &s1, &c1);
    const float e1 = expf(dAr);
    const float abr = e1 * c1, abi = e1 * s1;
#pragma unroll
    for (int j = 0; j < 16; ++j) {
      acc[j] += wr * pr - wi * pi;
      const float tr = pr * abr - pi * abi;
      pi = pr * abi + pi * abr;
      pr = tr;
    }
  }

  _Float16* row = Kpad + (size_t)h * KROW;
  if (t < KPAD) row[t] = (_Float16)0.0f;
#pragma unroll
  for (int j = 0; j < 16; ++j) row[KPAD + l0 + j] = (_Float16)acc[j];
}

// ---------------------------------------------------------------------------
// Kernel 2: depthwise causal conv via Toeplitz-tiled WMMA, fused D-skip +
// exact GELU. Block = one channel h, 8 waves. A[i,j] = K[L0-S0 + i - j],
// B[j,n] = u[n, S0+j] (n = batch; columns 8..15 zero). Output stored f16,
// transposed: Yt[(b*L + l)*512 + h]  -> GEMM B-fragments become b128 loads.
// ---------------------------------------------------------------------------
__global__ __launch_bounds__(256) void conv_skip_gelu(
    const _Float16* __restrict__ Kpad, const float* __restrict__ u,
    const float* __restrict__ Dskip, _Float16* __restrict__ Yt) {
  const int h = blockIdx.x;
  __shared__ _Float16 sK[KROW];
  __shared__ _Float16 sU[NBATCH][SEQ_L];

  const int t = threadIdx.x;
  for (int i = t; i < KROW; i += 256) sK[i] = Kpad[(size_t)h * KROW + i];
  for (int i = t; i < NBATCH * SEQ_L; i += 256) {
    const int bb = i >> 10, l = i & (SEQ_L - 1);
    sU[bb][l] = (_Float16)u[((size_t)bb * H_MODEL + h) * SEQ_L + l];
  }
  __syncthreads();

  const float dval = Dskip[h];
  const int wave = t >> 5, lane = t & 31;
  const int m = lane & 15;    // A-matrix row / B-matrix column / batch id
  const int hi = lane >> 4;   // K-half selector

  for (int ti = wave; ti < SEQ_L / 16; ti += 8) {
    const int L0 = ti * 16;
    v8f c = {};
    const int nchunks = ((L0 + 15) >> 5) + 1;
    for (int kc = 0; kc < nchunks; ++kc) {
      const int S0 = kc * 32;
      const int d0 = L0 - S0;
      v16h af, bf;
      // A: 16-bit 16x32 layout -> a[v] = A[m][ (v>>3)*16 + hi*8 + (v&7) ]
      const int base = KPAD + d0 + m - hi * 8;
#pragma unroll
      for (int w = 0; w < 8; ++w) {
        af[w]     = sK[base - w];        // taps g=0
        af[8 + w] = sK[base - 16 - w];   // taps g=1
      }
      if (m < NBATCH) {
        const v8h x0 = *(const v8h*)&sU[m][S0 + hi * 8];
        const v8h x1 = *(const v8h*)&sU[m][S0 + 16 + hi * 8];
#pragma unroll
        for (int w = 0; w < 8; ++w) { bf[w] = x0[w]; bf[8 + w] = x1[w]; }
      } else {
#pragma unroll
        for (int w = 0; w < 16; ++w) bf[w] = (_Float16)0.0f;
      }
      c = __builtin_amdgcn_wmma_f32_16x16x32_f16(
          /*neg_a=*/false, af, /*neg_b=*/false, bf,
          /*c_mod=*/(short)0, c, /*reuse_a=*/false, /*reuse_b=*/false);
    }
    // D tile: lane column n = m (batch), rows = hi*8 + r
    if (m < NBATCH) {
#pragma unroll
      for (int r = 0; r < 8; ++r) {
        const int l = L0 + hi * 8 + r;
        const float x = c[r] + dval * (float)sU[m][l];
        const float g = 0.5f * x * (1.0f + erff(x * 0.70710678118f));
        Yt[((size_t)m * SEQ_L + l) * H_MODEL + h] = (_Float16)g;
      }
    }
  }
}

// ---------------------------------------------------------------------------
// Kernel 3 helpers.
// ---------------------------------------------------------------------------
__global__ void f32_to_f16(const float* __restrict__ src,
                           _Float16* __restrict__ dst, int n) {
  const int i = blockIdx.x * 256 + threadIdx.x;
  if (i < n) dst[i] = (_Float16)src[i];
}

__global__ void zero_f32(float* __restrict__ p, int n) {
  const int i = blockIdx.x * 256 + threadIdx.x;
  if (i < n) p[i] = 0.0f;
}

// ---------------------------------------------------------------------------
// Kernel 3: Z = W(1024x512) x Y(512x8192) with fused bias + GLU + mean-pool.
// grid = (512 col-tiles, 8 row-groups), 4 waves/block; each wave computes the
// `a` tile (rows oa..oa+15) and paired `b` tile (rows oa+512..) -> 32 WMMAs.
// All fragments load as aligned 16-byte vectors. GLU tile is reduced over its
// 16 columns (one batch per tile) and atomically folded into pooled[b][o].
// ---------------------------------------------------------------------------
__global__ __launch_bounds__(128) void gemm_glu_pool(
    const _Float16* __restrict__ Wh, const _Float16* __restrict__ Yt,
    const float* __restrict__ conv_b, float* __restrict__ pooled) {
  const int no = blockIdx.x;                    // column tile (16 (b,l) pairs)
  const int wave = threadIdx.x >> 5, lane = threadIdx.x & 31;
  const int mo = blockIdx.y * 4 + wave;         // row tile of `a` (0..31)
  const int oa = mo * 16;
  const int n = lane & 15, hi = lane >> 4;
  const int c0 = no * 16;

  v8f ca = {}, cb = {};
  const _Float16* wrowA = Wh + (size_t)(oa + n) * H_MODEL;
  const _Float16* wrowB = Wh + (size_t)(oa + H_MODEL + n) * H_MODEL;
  const _Float16* yrow  = Yt + (size_t)(c0 + n) * H_MODEL;

  for (int k = 0; k < H_MODEL / 32; ++k) {
    const int off = k * 32 + hi * 8;
    __builtin_prefetch((const void*)(yrow + off + 32), 0, 1);  // next K-panel
    const v8h a0 = *(const v8h*)(wrowA + off);
    const v8h a1 = *(const v8h*)(wrowA + off + 16);
    const v8h b0 = *(const v8h*)(wrowB + off);
    const v8h b1 = *(const v8h*)(wrowB + off + 16);
    const v8h y0 = *(const v8h*)(yrow + off);
    const v8h y1 = *(const v8h*)(yrow + off + 16);
    v16h fa, fb, fy;
#pragma unroll
    for (int w = 0; w < 8; ++w) {
      fa[w] = a0[w]; fa[8 + w] = a1[w];
      fb[w] = b0[w]; fb[8 + w] = b1[w];
      fy[w] = y0[w]; fy[8 + w] = y1[w];
    }
    ca = __builtin_amdgcn_wmma_f32_16x16x32_f16(false, fa, false, fy,
                                                (short)0, ca, false, false);
    cb = __builtin_amdgcn_wmma_f32_16x16x32_f16(false, fb, false, fy,
                                                (short)0, cb, false, false);
  }

  const int bb = c0 >> 10;           // all 16 columns share one batch
  const float inv = 1.0f / (float)SEQ_L;
#pragma unroll
  for (int r = 0; r < 8; ++r) {
    const int o = oa + hi * 8 + r;
    const float av = ca[r] + conv_b[o];
    const float bv = cb[r] + conv_b[o + H_MODEL];
    float g = av * (1.0f / (1.0f + expf(-bv)));
    // sum the 16 columns: xor-shuffles stay inside each 16-lane half
    for (int s = 8; s > 0; s >>= 1) g += __shfl_xor(g, s, 32);
    if (n == 0) atomicAdd(&pooled[bb * H_MODEL + o], g * inv);
  }
}

// ---------------------------------------------------------------------------
// Kernel 4: out[b] = pooled[b,:] . dec_w + dec_b
// ---------------------------------------------------------------------------
__global__ __launch_bounds__(256) void decode_head(
    const float* __restrict__ pooled, const float* __restrict__ dec_w,
    const float* __restrict__ dec_b, float* __restrict__ out) {
  const int bb = blockIdx.x;
  __shared__ float red[256];
  float s = 0.0f;
  for (int o = threadIdx.x; o < H_MODEL; o += 256)
    s += pooled[bb * H_MODEL + o] * dec_w[o];
  red[threadIdx.x] = s;
  __syncthreads();
  for (int st = 128; st > 0; st >>= 1) {
    if (threadIdx.x < st) red[threadIdx.x] += red[threadIdx.x + st];
    __syncthreads();
  }
  if (threadIdx.x == 0) out[bb] = red[0] + dec_b[0];
}

// ---------------------------------------------------------------------------
// Launch: all on `stream`, graph-capture safe (no alloc/sync/events).
// ---------------------------------------------------------------------------
extern "C" void kernel_launch(void* const* d_in, const int* in_sizes, int n_in,
                              void* d_out, int out_size, void* d_ws,
                              size_t ws_size, hipStream_t stream) {
  (void)in_sizes; (void)n_in; (void)out_size; (void)ws_size;
  const float* u          = (const float*)d_in[0];
  const float* log_dt     = (const float*)d_in[1];
  const float* log_A_real = (const float*)d_in[2];
  const float* A_imag     = (const float*)d_in[3];
  const float* B_real     = (const float*)d_in[4];
  const float* B_imag     = (const float*)d_in[5];
  const float* C_real     = (const float*)d_in[6];
  const float* C_imag     = (const float*)d_in[7];
  const float* Dskip      = (const float*)d_in[8];
  const float* conv_w     = (const float*)d_in[9];
  const float* conv_b     = (const float*)d_in[10];
  const float* dec_w      = (const float*)d_in[11];
  const float* dec_b      = (const float*)d_in[12];
  float* out = (float*)d_out;

  // Workspace carve-up (256-byte aligned slices).
  char* ws = (char*)d_ws;
  size_t off = 0;
  auto take = [&](size_t bytes) {
    void* p = ws + off;
    off += (bytes + 255) & ~(size_t)255;
    return p;
  };
  _Float16* Kpad = (_Float16*)take((size_t)H_MODEL * KROW * 2);        // 1.06 MB
  _Float16* Wh   = (_Float16*)take((size_t)O_MODEL * H_MODEL * 2);     // 1 MB
  _Float16* Yt   = (_Float16*)take((size_t)NBATCH * SEQ_L * H_MODEL * 2); // 8 MB
  float*    pooled = (float*)take((size_t)NBATCH * H_MODEL * 4);       // 16 KB

  // 1) S4D kernel synthesis (one block per channel).
  s4d_make_kernel<<<H_MODEL, 64, 0, stream>>>(log_dt, log_A_real, A_imag,
                                              B_real, B_imag, C_real, C_imag,
                                              Kpad);
  // 2) Weight downconvert for WMMA.
  const int wn = O_MODEL * H_MODEL;
  f32_to_f16<<<(wn + 255) / 256, 256, 0, stream>>>(conv_w, Wh, wn);
  // 3) Depthwise causal conv + D-skip + GELU (WMMA Toeplitz).
  conv_skip_gelu<<<H_MODEL, 256, 0, stream>>>(Kpad, u, Dskip, Yt);
  // 4) Zero the pooling accumulator every launch (graph-replay safe).
  zero_f32<<<(NBATCH * H_MODEL + 255) / 256, 256, 0, stream>>>(
      pooled, NBATCH * H_MODEL);
  // 5) Pointwise GEMM + bias + GLU + mean-pool (WMMA).
  dim3 ggrid(NBATCH * SEQ_L / 16, 8);
  gemm_glu_pool<<<ggrid, 128, 0, stream>>>(Wh, Yt, conv_b, pooled);
  // 6) Decode head.
  decode_head<<<NBATCH, 256, 0, stream>>>(pooled, dec_w, dec_b, out);
}
